// SelfAttention_57217554317760
// MI455X (gfx1250) — matrix operand order repaired and checked by reading, not compiled, not run on hardware
//
#include <hip/hip_runtime.h>
#include <stdint.h>

#define H 16
#define D 64
#define DX 1024
#define BATCH 4
#define SEQ 2048
#define NTOK (BATCH * SEQ)   // 8192 tokens

typedef __attribute__((ext_vector_type(16))) __bf16         v16bf;
typedef __attribute__((ext_vector_type(8)))  float          v8f;
typedef __attribute__((ext_vector_type(4)))  unsigned int   v4u;
typedef __attribute__((ext_vector_type(8)))  int            v8i;
typedef __attribute__((ext_vector_type(4)))  int            v4i;
typedef __attribute__((ext_vector_type(4)))  unsigned short v4us;

#if __has_builtin(__builtin_amdgcn_tensor_load_to_lds)
#define HAVE_TDM 1
#else
#define HAVE_TDM 0
#endif

union BFrag { v4u q[2]; v16bf v; };   // 32 bytes: 16 bf16 = one WMMA A/B operand

__device__ __forceinline__ unsigned short f2bf(float f) {
  unsigned int u = __float_as_uint(f);
  unsigned int r = (u + 0x7FFFu + ((u >> 16) & 1u)) >> 16;   // RNE
  return (unsigned short)r;
}
__device__ __forceinline__ float bf2f(unsigned short h) {
  return __uint_as_float(((unsigned int)h) << 16);
}

__device__ __forceinline__ v8f wmma_bf16(const BFrag& a, const BFrag& b, v8f c) {
  return __builtin_amdgcn_wmma_f32_16x16x32_bf16(
      false, a.v, false, b.v, (short)0, c, false, false);
}

// CDNA5 async global->LDS copy (ASYNCcnt), 16 bytes per lane.
__device__ __forceinline__ void async_copy_b128(const unsigned short* gsrc, void* ldst) {
  unsigned ldsoff = (unsigned)(uintptr_t)ldst;            // addr[31:0] = LDS offset
  asm volatile("global_load_async_to_lds_b128 %0, %1, off"
               :: "v"(ldsoff), "v"((unsigned long long)(uintptr_t)gsrc)
               : "memory");
}
__device__ __forceinline__ void wait_async0() {
  asm volatile("s_wait_asynccnt 0x0" ::: "memory");
}
__device__ __forceinline__ void wait_tensor0() {
#if __has_builtin(__builtin_amdgcn_s_wait_tensorcnt)
  __builtin_amdgcn_s_wait_tensorcnt(0);
#else
  asm volatile("s_wait_tensorcnt 0x0" ::: "memory");
#endif
}

// ---------------------------------------------------------------------------
// fp32 -> bf16 cast, 4 elements / thread
// ---------------------------------------------------------------------------
__global__ void cvt_f32_to_bf16_v4(const float4* __restrict__ in,
                                   v4us* __restrict__ out, int n4) {
  int i = blockIdx.x * blockDim.x + threadIdx.x;
  if (i < n4) {
    float4 f = in[i];
    v4us o = { f2bf(f.x), f2bf(f.y), f2bf(f.z), f2bf(f.w) };
    out[i] = o;
  }
}

// ---------------------------------------------------------------------------
// GEMM: Y[M,N] = X[M,K](bf16) * W[N,K](bf16)^T + bias   (torch Linear)
// Block: 256 threads = 8 waves; block tile 256(M) x 64(N); wave = 32 rows x 64 cols.
// W tiles double-buffered in LDS via TDM (tensor_load_to_lds) or async copies.
// mode 0: bf16 into head layout [B,H,S,D]; mode 1: fp32 flat [M,N]
// ---------------------------------------------------------------------------
__global__ void gemm_bf16(const unsigned short* __restrict__ X,
                          const unsigned short* __restrict__ W,
                          const float* __restrict__ bias,
                          unsigned short* __restrict__ outBF,
                          float* __restrict__ outF32,
                          int mode) {
  __shared__ __align__(16) unsigned short Wt[2][64][32];   // [buf][col][k]  2 x 4 KB

  const int tid  = threadIdx.x;
  const int lane = tid & 31;
  const int wave = tid >> 5;
  const int n    = lane & 15;
  const int kh   = lane >> 4;
  const int rowBase = blockIdx.x * 256 + wave * 32;
  const int colBase = blockIdx.y * 64;
  const size_t xrow0 = (size_t)(rowBase + n) * DX;
  const size_t xrow1 = (size_t)(rowBase + 16 + n) * DX;

  // fallback cooperative staging assignment (16 B / thread)
  const int scol  = tid >> 2;
  const int skoff = (tid & 3) * 8;
  const size_t wbase = (size_t)(colBase + scol) * DX + skoff;

  auto stage = [&](int buf, int k0) {
#if HAVE_TDM
    if (wave == 0) {
      // D# group0: count=1, lds_addr, 57-bit global_addr, type=2
      const uintptr_t ga = (uintptr_t)&W[(size_t)colBase * DX + k0];
      const unsigned  la = (unsigned)(uintptr_t)&Wt[buf][0][0];
      v4u g0 = { 1u, la, (unsigned)ga,
                 (unsigned)((ga >> 32) & 0x01FFFFFFu) | 0x80000000u };
      // D# group1: data_size=1(2B); tensor 1024x1024; tile 32(k) x 64(col);
      //            tensor_dim0_stride=1024
      v8i g1 = { (int)(1u << 16),
                 (int)((DX & 0xFFFFu) << 16),                    // dim0[15:0]
                 (int)(((DX >> 16) & 0xFFFFu) | ((DX & 0xFFFFu) << 16)), // dim0 hi | dim1 lo
                 (int)(((DX >> 16) & 0xFFFFu) | (32u << 16)),    // dim1 hi | tile_dim0
                 (int)64,                                        // tile_dim1 | tile_dim2=0
                 (int)DX, 0, 0 };                                // dim0_stride
      v4i z4 = { 0, 0, 0, 0 };
      v8i z8 = { 0, 0, 0, 0, 0, 0, 0, 0 };
      __builtin_amdgcn_tensor_load_to_lds(g0, g1, z4, z4, z8, 0);
    }
#else
    async_copy_b128(&W[wbase + k0], &Wt[buf][scol][skoff]);
#endif
  };
  auto stage_wait = [&]() {
#if HAVE_TDM
    if (wave == 0) wait_tensor0();
#else
    wait_async0();
#endif
  };

  BFrag a_cur[2], a_nxt[2];
  a_cur[0].q[0] = *(const v4u*)&X[xrow0 + kh * 8];
  a_cur[0].q[1] = *(const v4u*)&X[xrow0 + kh * 8 + 16];
  a_cur[1].q[0] = *(const v4u*)&X[xrow1 + kh * 8];
  a_cur[1].q[1] = *(const v4u*)&X[xrow1 + kh * 8 + 16];
  a_nxt[0] = a_cur[0]; a_nxt[1] = a_cur[1];
  stage(0, 0);
  stage_wait();
  __syncthreads();

  v8f acc[2][4] = {};
  for (int k0 = 0; k0 < DX; k0 += 32) {
    const int buf = (k0 >> 5) & 1;
    BFrag bb[4];                       // preload all B fragments, then WMMA burst
#pragma unroll
    for (int nt = 0; nt < 4; ++nt) {
      bb[nt].q[0] = *(const v4u*)&Wt[buf][nt * 16 + n][kh * 8];
      bb[nt].q[1] = *(const v4u*)&Wt[buf][nt * 16 + n][kh * 8 + 16];
    }
    if (k0 + 32 < DX) {
      stage(buf ^ 1, k0 + 32);
      __builtin_prefetch(&X[xrow0 + k0 + 64], 0, 3);
      a_nxt[0].q[0] = *(const v4u*)&X[xrow0 + k0 + 32 + kh * 8];
      a_nxt[0].q[1] = *(const v4u*)&X[xrow0 + k0 + 32 + kh * 8 + 16];
      a_nxt[1].q[0] = *(const v4u*)&X[xrow1 + k0 + 32 + kh * 8];
      a_nxt[1].q[1] = *(const v4u*)&X[xrow1 + k0 + 32 + kh * 8 + 16];
    }
#pragma unroll
    for (int nt = 0; nt < 4; ++nt) {
      acc[0][nt] = wmma_bf16(a_cur[0], bb[nt], acc[0][nt]);
      acc[1][nt] = wmma_bf16(a_cur[1], bb[nt], acc[1][nt]);
    }
    a_cur[0] = a_nxt[0]; a_cur[1] = a_nxt[1];
    stage_wait();
    __syncthreads();
  }

#pragma unroll
  for (int rg = 0; rg < 2; ++rg) {
#pragma unroll
    for (int nt = 0; nt < 4; ++nt) {
      const int col = colBase + nt * 16 + n;
      const float bv = bias[col];
#pragma unroll
      for (int v = 0; v < 8; ++v) {
        const int row = rowBase + rg * 16 + v + 8 * kh;
        const float val = acc[rg][nt][v] + bv;
        if (mode == 0) {
          const int bb2 = row >> 11;
          const int s   = row & (SEQ - 1);
          const int hh  = col >> 6;
          const int d   = col & (D - 1);
          outBF[(((size_t)bb2 * H + hh) * SEQ + s) * D + d] = f2bf(val);
        } else {
          outF32[(size_t)row * DX + col] = val;
        }
      }
    }
  }
}

// ---------------------------------------------------------------------------
// Flash attention per head; K/V tiles double-buffered, K via async LDS copies.
// grid (SEQ/128, H, B), 256 threads = 8 waves; wave owns 16 query rows x D=64.
// ---------------------------------------------------------------------------
__global__ void flash_attn(const unsigned short* __restrict__ Qg,
                           const unsigned short* __restrict__ Kg,
                           const unsigned short* __restrict__ Vg,
                           const unsigned short* __restrict__ Rg,
                           unsigned short* __restrict__ AO) {
  __shared__ __align__(16) unsigned short Ktile[2][32][D];   // 2 x 4 KB
  __shared__ __align__(16) unsigned short Vt[2][D][32];      // 2 x 4 KB
  __shared__ __align__(16) float          Sbuf[8][16][32];
  __shared__ __align__(16) unsigned short Pbuf[8][16][32];
  __shared__ float alphaS[8][16];
  __shared__ float lsumS[8][16];

  const int tid  = threadIdx.x;
  const int lane = tid & 31;
  const int wave = tid >> 5;
  const int n    = lane & 15;
  const int kh   = lane >> 4;
  const size_t headBase = ((size_t)blockIdx.z * H + blockIdx.y) * SEQ * D;
  const int qrow = blockIdx.x * 128 + wave * 16 + n;

  const int sj = tid >> 3;           // 0..31
  const int sd = (tid & 7) * 8;      // 0..56

  BFrag qf[2];
#pragma unroll
  for (int ks = 0; ks < 2; ++ks) {
    const size_t qoff = headBase + (size_t)qrow * D + ks * 32 + kh * 8;
    qf[ks].q[0] = *(const v4u*)&Qg[qoff];
    qf[ks].q[1] = *(const v4u*)&Qg[qoff + 16];
  }

  v8f Oacc[4] = {};
  float m_row = -__builtin_inff();
  float l_row = 0.f;
  const float SCALE = 0.125f * 1.44269504088896f;   // 1/sqrt(D) * log2(e)

  // prologue: stage chunk 0
  async_copy_b128(&Kg[headBase + (size_t)sj * D + sd], &Ktile[0][sj][sd]);
  {
    v4u vv = *(const v4u*)&Vg[headBase + (size_t)sj * D + sd];
    const unsigned short* pv = (const unsigned short*)&vv;
#pragma unroll
    for (int i = 0; i < 8; ++i) Vt[0][sd + i][sj] = pv[i];
  }
  wait_async0();
  __syncthreads();

  for (int j0 = 0; j0 < SEQ; j0 += 32) {
    const int buf = (j0 >> 5) & 1;
    const bool has_next = (j0 + 32) < SEQ;
    v4u vnext = {};
    if (has_next) {
      const size_t g = headBase + (size_t)(j0 + 32 + sj) * D + sd;
      async_copy_b128(&Kg[g], &Ktile[buf ^ 1][sj][sd]);   // overlaps with compute
      vnext = *(const v4u*)&Vg[g];
    }

    // S = Q K^T : preload all K fragments, then WMMA burst
    BFrag kb[2][2];
#pragma unroll
    for (int ks = 0; ks < 2; ++ks) {
#pragma unroll
      for (int nt = 0; nt < 2; ++nt) {
        kb[ks][nt].q[0] = *(const v4u*)&Ktile[buf][nt * 16 + n][ks * 32 + kh * 8];
        kb[ks][nt].q[1] = *(const v4u*)&Ktile[buf][nt * 16 + n][ks * 32 + kh * 8 + 16];
      }
    }
    v8f s0 = {}, s1 = {};
#pragma unroll
    for (int ks = 0; ks < 2; ++ks) {
      s0 = wmma_bf16(qf[ks], kb[ks][0], s0);
      s1 = wmma_bf16(qf[ks], kb[ks][1], s1);
    }
#pragma unroll
    for (int v = 0; v < 8; ++v) {
      Sbuf[wave][v + 8 * kh][n]      = s0[v];
      Sbuf[wave][v + 8 * kh][16 + n] = s1[v];
    }
    __syncthreads();

    // online softmax: lane handles row n, columns [kh*16, kh*16+16)
    float rmax = -__builtin_inff();
#pragma unroll
    for (int c = 0; c < 16; ++c)
      rmax = fmaxf(rmax, Sbuf[wave][n][kh * 16 + c] * SCALE);
    rmax = fmaxf(rmax, __shfl_xor(rmax, 16, 32));
    const float m_new = fmaxf(m_row, rmax);
    const float alpha = exp2f(m_row - m_new);
    float psum = 0.f;
#pragma unroll
    for (int c = 0; c < 16; ++c) {
      const float p = exp2f(Sbuf[wave][n][kh * 16 + c] * SCALE - m_new);
      psum += p;
      Pbuf[wave][n][kh * 16 + c] = f2bf(p);
    }
    psum += __shfl_xor(psum, 16, 32);
    l_row = l_row * alpha + psum;
    m_row = m_new;
    if (lane < 16) alphaS[wave][n] = alpha;
    __syncthreads();

    // write next V tile transposed (buf^1 free: all reads finished last iter)
    if (has_next) {
      const unsigned short* pv = (const unsigned short*)&vnext;
#pragma unroll
      for (int i = 0; i < 8; ++i) Vt[buf ^ 1][sd + i][sj] = pv[i];
    }

    float av[8];
#pragma unroll
    for (int v = 0; v < 8; ++v) av[v] = alphaS[wave][v + 8 * kh];

    BFrag pf;
    pf.q[0] = *(const v4u*)&Pbuf[wave][n][kh * 8];
    pf.q[1] = *(const v4u*)&Pbuf[wave][n][kh * 8 + 16];
    BFrag vf[4];                        // preload all V fragments
#pragma unroll
    for (int dt = 0; dt < 4; ++dt) {
      vf[dt].q[0] = *(const v4u*)&Vt[buf][dt * 16 + n][kh * 8];
      vf[dt].q[1] = *(const v4u*)&Vt[buf][dt * 16 + n][kh * 8 + 16];
    }
#pragma unroll
    for (int dt = 0; dt < 4; ++dt) {
#pragma unroll
      for (int v = 0; v < 8; ++v) Oacc[dt][v] *= av[v];
      Oacc[dt] = wmma_bf16(pf, vf[dt], Oacc[dt]);
    }

    wait_async0();                      // next K tile landed
    __syncthreads();                    // all waves done with this buffer
  }

  if (lane < 16) lsumS[wave][n] = l_row;
  __syncthreads();

  const int s_base = blockIdx.x * 128 + wave * 16;
#pragma unroll
  for (int v = 0; v < 8; ++v) {
    const int r = v + 8 * kh;
    const float inv = 1.0f / lsumS[wave][r];
    const int srow = s_base + r;
#pragma unroll
    for (int dt = 0; dt < 4; ++dt) {
      const int d = dt * 16 + n;
      const float rg = bf2f(Rg[headBase + (size_t)srow * D + d]);
      const float o = Oacc[dt][v] * inv * rg;
      AO[((size_t)blockIdx.z * SEQ + srow) * DX + blockIdx.y * D + d] = f2bf(o);
    }
  }
}

// ---------------------------------------------------------------------------
extern "C" void kernel_launch(void* const* d_in, const int* in_sizes, int n_in,
                              void* d_out, int out_size, void* d_ws, size_t ws_size,
                              hipStream_t stream) {
  const float* value = (const float*)d_in[0];
  const float* key_  = (const float*)d_in[1];
  const float* query = (const float*)d_in[2];
  const float* Wq_w  = (const float*)d_in[3];  const float* Wq_b = (const float*)d_in[4];
  const float* Wk_w  = (const float*)d_in[5];  const float* Wk_b = (const float*)d_in[6];
  const float* Wv_w  = (const float*)d_in[7];  const float* Wv_b = (const float*)d_in[8];
  const float* Wr_w  = (const float*)d_in[9];  const float* Wr_b = (const float*)d_in[10];
  const float* Wo_w  = (const float*)d_in[11]; const float* Wo_b = (const float*)d_in[12];
  float* out = (float*)d_out;

  unsigned short* ws = (unsigned short*)d_ws;
  const size_t NX = (size_t)NTOK * DX;   // 8M elems
  const size_t NW = (size_t)DX * DX;     // 1M elems
  unsigned short* Xq = ws; ws += NX;
  unsigned short* Xk = ws; ws += NX;
  unsigned short* Xv = ws; ws += NX;
  unsigned short* Wq = ws; ws += NW;
  unsigned short* Wk = ws; ws += NW;
  unsigned short* Wv = ws; ws += NW;
  unsigned short* Wr = ws; ws += NW;
  unsigned short* Wo = ws; ws += NW;
  unsigned short* Qp = ws; ws += NX;     // [B,H,S,D]
  unsigned short* Kp = ws; ws += NX;
  unsigned short* Vp = ws; ws += NX;
  unsigned short* Rp = ws; ws += NX;
  unsigned short* AO = ws; ws += NX;     // [B,S,DX]

  const int T = 256;
  const int nx4 = (int)(NX / 4), nw4 = (int)(NW / 4);
  cvt_f32_to_bf16_v4<<<(nx4 + T - 1) / T, T, 0, stream>>>((const float4*)query, (v4us*)Xq, nx4);
  cvt_f32_to_bf16_v4<<<(nx4 + T - 1) / T, T, 0, stream>>>((const float4*)key_,  (v4us*)Xk, nx4);
  cvt_f32_to_bf16_v4<<<(nx4 + T - 1) / T, T, 0, stream>>>((const float4*)value, (v4us*)Xv, nx4);
  cvt_f32_to_bf16_v4<<<(nw4 + T - 1) / T, T, 0, stream>>>((const float4*)Wq_w, (v4us*)Wq, nw4);
  cvt_f32_to_bf16_v4<<<(nw4 + T - 1) / T, T, 0, stream>>>((const float4*)Wk_w, (v4us*)Wk, nw4);
  cvt_f32_to_bf16_v4<<<(nw4 + T - 1) / T, T, 0, stream>>>((const float4*)Wv_w, (v4us*)Wv, nw4);
  cvt_f32_to_bf16_v4<<<(nw4 + T - 1) / T, T, 0, stream>>>((const float4*)Wr_w, (v4us*)Wr, nw4);
  cvt_f32_to_bf16_v4<<<(nw4 + T - 1) / T, T, 0, stream>>>((const float4*)Wo_w, (v4us*)Wo, nw4);

  dim3 gg(NTOK / 256, DX / 64);
  gemm_bf16<<<gg, 256, 0, stream>>>(Xq, Wq, Wq_b, Qp, nullptr, 0);
  gemm_bf16<<<gg, 256, 0, stream>>>(Xk, Wk, Wk_b, Kp, nullptr, 0);
  gemm_bf16<<<gg, 256, 0, stream>>>(Xv, Wv, Wv_b, Vp, nullptr, 0);
  gemm_bf16<<<gg, 256, 0, stream>>>(Xq, Wr, Wr_b, Rp, nullptr, 0);

  dim3 ga(SEQ / 128, H, BATCH);
  flash_attn<<<ga, 256, 0, stream>>>(Qp, Kp, Vp, Rp, AO);

  gemm_bf16<<<gg, 256, 0, stream>>>(AO, Wo, Wo_b, nullptr, out, 1);
}
